// MyModel_67259187855736
// MI455X (gfx1250) — matrix operand order repaired and checked by reading, not compiled, not run on hardware
//
#include <hip/hip_runtime.h>
#include <hip/hip_bf16.h>
#include <math.h>
#include <stdint.h>

// ---------------------------------------------------------------- types
typedef __attribute__((ext_vector_type(16))) __bf16 bf16x16;
typedef __attribute__((ext_vector_type(8)))  __bf16 bf16x8;
typedef __attribute__((ext_vector_type(8)))  float  f32x8;

#define NB   8
#define N0   784
#define NE   8192
#define DD   2048
#define NK1  471
#define NK2  283
#define NK3  170
#define MPADMAX 896   // 784 rounded up to 128

// ---------------------------------------------------------------- WMMA GEMM
// C[b][M][2048] = A[b][M][2048] @ W[2048][2048]
// A: bf16, zero-padded to Mpad rows.  Wt: bf16, pre-transposed to [N][K].
#define BM 128
#define BN 128
#define BK 32
#define LAS 40   // padded LDS row stride in bf16 elems (80B, 16B aligned)

__device__ __forceinline__ void async_b128(const void* g, const void* l) {
  uint32_t lo = (uint32_t)(uintptr_t)l;   // low 32 bits of flat LDS ptr == LDS offset
  asm volatile("global_load_async_to_lds_b128 %0, %1, off"
               :: "v"(lo), "v"(g) : "memory");
}

__global__ void __launch_bounds__(256)
k_gemm_wmma(const __bf16* __restrict__ A, const __bf16* __restrict__ Wt,
            float* __restrict__ C, int M, long bsA, long bsC)
{
  const int K = DD, NN = DD;
  __shared__ __bf16 la[2][BM * LAS];
  __shared__ __bf16 lb[2][BN * LAS];

  const int b    = blockIdx.z;
  const __bf16* Ab = A + (long)b * bsA;
  float*        Cb = C + (long)b * bsC;
  const int m0   = blockIdx.y * BM;
  const int n0   = blockIdx.x * BN;
  const int tid  = threadIdx.x;
  const int lane = tid & 31;
  const int w    = tid >> 5;
  const int wm   = w >> 2;   // 0..1 -> 64-row slab
  const int wn   = w & 3;    // 0..3 -> 32-col slab
  const int l15  = lane & 15;
  const int hiA  = (lane < 16) ? 0 : 8;    // A frag K base
  const int koB  = (lane < 16) ? 0 : 8;    // B frag K base

  f32x8 acc[4][2];
#pragma unroll
  for (int i = 0; i < 4; ++i)
#pragma unroll
    for (int j = 0; j < 2; ++j)
      acc[i][j] = (f32x8){0.f,0.f,0.f,0.f,0.f,0.f,0.f,0.f};

  // per-thread: 2 A chunks + 2 B chunks per tile -> 4 async instrs / wave / tile
  auto issue = [&](int bufi, int kk) {
#pragma unroll
    for (int i = 0; i < 2; ++i) {
      int li  = tid + i * 256;        // 0..511 of 512 16B-chunks
      int row = li >> 2;              // 128 rows, 4 chunks each
      int j   = (li & 3) << 3;        // bf16 offset within row (0,8,16,24)
      async_b128(Ab + (long)(m0 + row) * K + kk + j, &la[bufi][row * LAS + j]);
      async_b128(Wt + (long)(n0 + row) * K + kk + j, &lb[bufi][row * LAS + j]);
    }
  };

  const int NT = K / BK;   // 64
  issue(0, 0);
  issue(1, BK);

  int bufi = 0;
  for (int it = 0; it < NT; ++it) {
    if (it + 1 < NT) asm volatile("s_wait_asynccnt 0x4" ::: "memory");
    else             asm volatile("s_wait_asynccnt 0x0" ::: "memory");
    __syncthreads();                       // tile `it` visible from all waves

    bf16x16 afr[4], bfr[2];
#pragma unroll
    for (int i = 0; i < 4; ++i) {
      const __bf16* p = &la[bufi][(wm * 64 + i * 16 + l15) * LAS + hiA];
      union { bf16x16 v; bf16x8 h[2]; } u;
      u.h[0] = *(const bf16x8*)p;          // K = hiA .. hiA+7
      u.h[1] = *(const bf16x8*)(p + 16);   // K = hiA+16 .. hiA+23
      afr[i] = u.v;
    }
#pragma unroll
    for (int j = 0; j < 2; ++j) {
      const __bf16* p = &lb[bufi][(wn * 32 + j * 16 + l15) * LAS + koB];
      union { bf16x16 v; bf16x8 h[2]; } u;
      u.h[0] = *(const bf16x8*)p;          // K = koB .. koB+7
      u.h[1] = *(const bf16x8*)(p + 16);   // K = koB+16 .. koB+23
      bfr[j] = u.v;
    }
    asm volatile("s_wait_dscnt 0x0" ::: "memory");  // frags in regs, LDS reads done
    __syncthreads();                                 // safe to overwrite buf

    if (it + 2 < NT) issue(bufi, (it + 2) * BK);     // overlap fill with math

#pragma unroll
    for (int i = 0; i < 4; ++i)
#pragma unroll
      for (int j = 0; j < 2; ++j)
        acc[i][j] = __builtin_amdgcn_wmma_f32_16x16x32_bf16(
            false, afr[i], false, bfr[j], (short)0, acc[i][j], false, false);
    bufi ^= 1;
  }

  // epilogue: C/D layout -> VGPR r: lanes0-15 (M=r, N=lane), lanes16-31 (M=r+8)
#pragma unroll
  for (int i = 0; i < 4; ++i) {
    int rbase = m0 + wm * 64 + i * 16 + ((lane < 16) ? 0 : 8);
#pragma unroll
    for (int j = 0; j < 2; ++j) {
      int c = n0 + wn * 32 + j * 16 + l15;
      union { f32x8 v; float f[8]; } u; u.v = acc[i][j];
#pragma unroll
      for (int r = 0; r < 8; ++r) {
        int grow = rbase + r;
        if (grow < M) Cb[(long)grow * NN + c] = u.f[r];
      }
    }
  }
}

// ---------------------------------------------------------------- precision prep
// A: fp32 [b][n][DD] -> bf16 [b][MPADMAX][DD], rows n..Mpad zero-filled
__global__ void k_convA(__bf16* __restrict__ out, const float* __restrict__ hin,
                        int n, int Mpad, long bsIn) {
  int b = blockIdx.y;
  long idx = (long)blockIdx.x * 256 + threadIdx.x;
  long tot = (long)Mpad * DD;
  if (idx < tot) {
    int row = (int)(idx >> 11);
    float v = (row < n) ? hin[(long)b * bsIn + idx] : 0.f;
    out[(long)b * MPADMAX * DD + idx] = (__bf16)v;
  }
}

// W: fp32 [K][N] -> bf16 [N][K] (tiled transpose through LDS)
__global__ void __launch_bounds__(256)
k_convW(const float* __restrict__ Wsrc, __bf16* __restrict__ Wt) {
  __shared__ __bf16 t[32][33];
  int k0 = blockIdx.y * 32;
  int n0 = blockIdx.x * 32;
  int tx = threadIdx.x & 31, ty = threadIdx.x >> 5;
  for (int r = ty; r < 32; r += 8)
    t[r][tx] = (__bf16)Wsrc[(long)(k0 + r) * DD + n0 + tx];
  __syncthreads();
  for (int r = ty; r < 32; r += 8)
    Wt[(long)(n0 + r) * DD + k0 + tx] = t[tx][r];
}

// ---------------------------------------------------------------- small kernels
__global__ void k_fill(float* __restrict__ p, long count, float v) {
  long i = (long)blockIdx.x * blockDim.x + threadIdx.x;
  if (i < count) p[i] = v;
}

__global__ void k_init_edges(const int* __restrict__ ei, int* __restrict__ s,
                             int* __restrict__ d, float* __restrict__ m) {
  int b = blockIdx.y; int e = blockIdx.x * 256 + threadIdx.x;
  if (e < NE) {
    s[b * NE + e] = ei[(long)b * 2 * NE + e];
    d[b * NE + e] = ei[(long)b * 2 * NE + NE + e];
    m[b * NE + e] = 1.0f;
  }
}

__global__ void k_edge_deg(float* __restrict__ deg, const int* __restrict__ dst,
                           const float* __restrict__ m) {
  int b = blockIdx.y; int e = blockIdx.x * 256 + threadIdx.x;
  if (e < NE) atomicAdd(&deg[b * N0 + dst[b * NE + e]], m[b * NE + e]);
}

__global__ void k_coef(float* __restrict__ coef, const int* __restrict__ src,
                       const int* __restrict__ dst, const float* __restrict__ m,
                       const float* __restrict__ deg) {
  int b = blockIdx.y; int e = blockIdx.x * 256 + threadIdx.x;
  if (e < NE) {
    float mv = m[b * NE + e];
    coef[b * NE + e] = mv * rsqrtf(deg[b * N0 + src[b * NE + e]])
                          * rsqrtf(deg[b * N0 + dst[b * NE + e]]);
  }
}

__global__ void k_agg(float* __restrict__ agg, const float* __restrict__ xw,
                      const float* __restrict__ coef, const int* __restrict__ src,
                      const int* __restrict__ dst, long bs) {
  int e = blockIdx.x, b = blockIdx.y;
  float c = coef[b * NE + e];
  if (c != 0.f) {
    const float* hs = xw  + (long)b * bs + (long)src[b * NE + e] * DD;
    float*       ad = agg + (long)b * bs + (long)dst[b * NE + e] * DD;
    for (int d0 = threadIdx.x; d0 < DD; d0 += 256) atomicAdd(&ad[d0], c * hs[d0]);
  }
}

__global__ void k_combine(float* __restrict__ agg, const float* __restrict__ xw,
                          const float* __restrict__ deg, const float* __restrict__ bias,
                          int n, long bs) {
  int b = blockIdx.y;
  long idx = (long)blockIdx.x * 256 + threadIdx.x;
  long tot = (long)n * DD;
  if (idx < tot) {
    int i = (int)(idx >> 11); int d = (int)(idx & (DD - 1));
    float v = agg[(long)b * bs + idx] + xw[(long)b * bs + idx] / deg[b * N0 + i] + bias[d];
    agg[(long)b * bs + idx] = fmaxf(v, 0.f);
  }
}

__global__ void k_pnorm(const float* __restrict__ p, float* __restrict__ rn) {
  __shared__ float red[256];
  float s = 0.f;
  for (int d = threadIdx.x; d < DD; d += 256) s += p[d] * p[d];
  red[threadIdx.x] = s; __syncthreads();
  for (int o = 128; o > 0; o >>= 1) {
    if (threadIdx.x < o) red[threadIdx.x] += red[threadIdx.x + o];
    __syncthreads();
  }
  if (threadIdx.x == 0) rn[0] = rsqrtf(red[0]);
}

__global__ void k_score(float* __restrict__ score, const float* __restrict__ h,
                        const float* __restrict__ p, const float* __restrict__ rn,
                        long bs) {
  __shared__ float red[256];
  int b = blockIdx.y, i = blockIdx.x;
  const float* hr = h + (long)b * bs + (long)i * DD;
  float s = 0.f;
  for (int d = threadIdx.x; d < DD; d += 256) s += hr[d] * p[d];
  red[threadIdx.x] = s; __syncthreads();
  for (int o = 128; o > 0; o >>= 1) {
    if (threadIdx.x < o) red[threadIdx.x] += red[threadIdx.x + o];
    __syncthreads();
  }
  if (threadIdx.x == 0) score[b * N0 + i] = red[0] * rn[0];
}

// per-graph top-k via 1024-wide bitonic sort (descending) in LDS
__global__ void __launch_bounds__(512)
k_topk(const float* __restrict__ score, float* __restrict__ vals,
       int* __restrict__ perm, int* __restrict__ nidx, int n, int kout) {
  __shared__ float ss[1024];
  __shared__ int   ii[1024];
  int b = blockIdx.x;
  for (int i = threadIdx.x; i < 1024; i += 512) {
    ss[i] = (i < n) ? score[b * N0 + i] : -3.4e38f;
    ii[i] = i;
  }
  __syncthreads();
  for (int k = 2; k <= 1024; k <<= 1) {
    for (int j = k >> 1; j > 0; j >>= 1) {
      for (int t = threadIdx.x; t < 1024; t += 512) {
        int ixj = t ^ j;
        if (ixj > t) {
          bool up = ((t & k) == 0);
          float a = ss[t], c = ss[ixj];
          bool sw = up ? (a < c) : (a > c);   // overall descending
          if (sw) {
            ss[t] = c; ss[ixj] = a;
            int tmp = ii[t]; ii[t] = ii[ixj]; ii[ixj] = tmp;
          }
        }
      }
      __syncthreads();
    }
  }
  for (int i = threadIdx.x; i < n; i += 512) nidx[b * N0 + i] = -1;
  __syncthreads();
  for (int t = threadIdx.x; t < kout; t += 512) {
    perm[b * NK1 + t] = ii[t];
    vals[b * NK1 + t] = ss[t];
    nidx[b * N0 + ii[t]] = t;
  }
}

__global__ void k_gather(float* __restrict__ out, const float* __restrict__ hin,
                         const int* __restrict__ perm, const float* __restrict__ vals,
                         long bsOut, long bsIn) {
  int b = blockIdx.y, t = blockIdx.x;
  int sr = perm[b * NK1 + t];
  float sc = tanhf(vals[b * NK1 + t]);
  const float* s = hin + (long)b * bsIn  + (long)sr * DD;
  float*       d = out + (long)b * bsOut + (long)t  * DD;
  for (int i = threadIdx.x; i < DD; i += 256) d[i] = s[i] * sc;
}

__global__ void k_remap(int* __restrict__ so, int* __restrict__ dno, float* __restrict__ mo,
                        const int* __restrict__ si, const int* __restrict__ di,
                        const float* __restrict__ mi, const int* __restrict__ nidx) {
  int b = blockIdx.y; int e = blockIdx.x * 256 + threadIdx.x;
  if (e < NE) {
    int s0 = si[b * NE + e], d0 = di[b * NE + e];
    float mv = mi[b * NE + e];
    int s2 = nidx[b * N0 + s0], d2 = nidx[b * N0 + d0];
    bool valid = (s2 >= 0) && (d2 >= 0) && (mv > 0.f);
    so[b * NE + e]  = valid ? s2 : 0;
    dno[b * NE + e] = valid ? d2 : 0;
    mo[b * NE + e]  = valid ? 1.f : 0.f;
  }
}

__global__ void k_mean(float* __restrict__ g, const float* __restrict__ h,
                       int n, long bs) {
  int b = blockIdx.y; int d = blockIdx.x * 256 + threadIdx.x;
  const float* hb = h + (long)b * bs;
  float s = 0.f;
  for (int i = 0; i < n; ++i) s += hb[(long)i * DD + d];
  g[b * DD + d] = s / (float)n;
}

// mode 0: relu, mode 1: sigmoid
__global__ void k_fc(float* __restrict__ out, const float* __restrict__ act,
                     const float* __restrict__ wt, const float* __restrict__ bias,
                     int K, int N, int mode) {
  int b = blockIdx.y;
  int j = blockIdx.x * blockDim.x + threadIdx.x;
  if (j >= N) return;
  const float* a = act + (long)b * K;
  float s = bias[j];
  for (int k = 0; k < K; ++k) s += a[k] * wt[(long)k * N + j];
  s = (mode == 0) ? fmaxf(s, 0.f) : 1.f / (1.f + expf(-s));
  out[(long)b * N + j] = s;
}

// ---------------------------------------------------------------- launch
extern "C" void kernel_launch(void* const* d_in, const int* in_sizes, int n_in,
                              void* d_out, int out_size, void* d_ws, size_t ws_size,
                              hipStream_t stream) {
  (void)in_sizes; (void)n_in; (void)out_size; (void)ws_size;
  const float* x  = (const float*)d_in[0];
  const int*   ei = (const int*)d_in[1];
  const float* Ws[4] = {(const float*)d_in[2], (const float*)d_in[4],
                        (const float*)d_in[6], (const float*)d_in[8]};
  const float* bv[4] = {(const float*)d_in[3], (const float*)d_in[5],
                        (const float*)d_in[7], (const float*)d_in[9]};
  const float* ps[3] = {(const float*)d_in[10], (const float*)d_in[11], (const float*)d_in[12]};
  const float* lw1 = (const float*)d_in[13]; const float* lb1 = (const float*)d_in[14];
  const float* lw2 = (const float*)d_in[15]; const float* lb2 = (const float*)d_in[16];
  const float* lw3 = (const float*)d_in[17]; const float* lb3 = (const float*)d_in[18];

  char* w = (char*)d_ws; size_t off = 0;
  auto take = [&](size_t bytes) -> char* {
    char* r = w + off; off = (off + bytes + 255) & ~(size_t)255; return r;
  };
  float*  BUF1 = (float*) take((size_t)NB * N0  * DD * 4);  // xw
  float*  BUF2 = (float*) take((size_t)NB * N0  * DD * 4);  // agg / layer output
  float*  BUF0 = (float*) take((size_t)NB * NK1 * DD * 4);  // pooled features
  __bf16* Abf  = (__bf16*)take((size_t)NB * MPADMAX * DD * 2);
  __bf16* Wt   = (__bf16*)take((size_t)DD * DD * 2);
  int*   es[2]; int* ed[2]; float* em[2];
  for (int i = 0; i < 2; ++i) {
    es[i] = (int*)take((size_t)NB * NE * 4);
    ed[i] = (int*)take((size_t)NB * NE * 4);
    em[i] = (float*)take((size_t)NB * NE * 4);
  }
  float* deg   = (float*)take((size_t)NB * N0 * 4);
  float* coef  = (float*)take((size_t)NB * NE * 4);
  float* score = (float*)take((size_t)NB * N0 * 4);
  float* vals  = (float*)take((size_t)NB * NK1 * 4);
  int*   perm  = (int*)  take((size_t)NB * NK1 * 4);
  int*   nidx  = (int*)  take((size_t)NB * N0 * 4);
  float* g     = (float*)take((size_t)NB * DD * 4);
  float* mh1   = (float*)take((size_t)NB * 1024 * 4);
  float* mh2   = (float*)take((size_t)NB * 1024 * 4);
  float* rnorm = (float*)take(4);

  const int nin[4]  = {N0, NK1, NK2, NK3};
  const int kout[3] = {NK1, NK2, NK3};
  const long bsBig  = (long)N0  * DD;
  const long bsSm   = (long)NK1 * DD;
  const dim3 egrid((NE + 255) / 256, NB);

  int cur = 0;
  k_init_edges<<<egrid, 256, 0, stream>>>(ei, es[0], ed[0], em[0]);

  for (int l = 0; l < 4; ++l) {
    const int n = nin[l];
    const int Mpad = (n + BM - 1) & ~(BM - 1);
    const float* hin = (l == 0) ? x : BUF0;
    const long bsIn  = (l == 0) ? bsBig : bsSm;

    // bf16 prep: A (zero-padded) and W (transposed)
    k_convA<<<dim3((unsigned)(((long)Mpad * DD + 255) / 256), NB), 256, 0, stream>>>(
        Abf, hin, n, Mpad, bsIn);
    k_convW<<<dim3(DD / 32, DD / 32), 256, 0, stream>>>(Ws[l], Wt);

    // h = x @ W  (bf16 WMMA, async LDS fill, fp32 accumulate)
    k_gemm_wmma<<<dim3(DD / BN, Mpad / BM, NB), 256, 0, stream>>>(
        Abf, Wt, BUF1, n, (long)MPADMAX * DD, bsBig);

    // deg = 1 + segment_sum(m, dst)
    k_fill<<<(NB * N0 + 255) / 256, 256, 0, stream>>>(deg, (long)NB * N0, 1.0f);
    k_edge_deg<<<egrid, 256, 0, stream>>>(deg, ed[cur], em[cur]);
    k_coef<<<egrid, 256, 0, stream>>>(coef, es[cur], ed[cur], em[cur], deg);

    // agg = segment_sum(coef * h[src], dst)
    k_fill<<<(int)(((long)NB * N0 * DD + 255) / 256), 256, 0, stream>>>(
        BUF2, (long)NB * N0 * DD, 0.0f);
    k_agg<<<dim3(NE, NB), 256, 0, stream>>>(BUF2, BUF1, coef, es[cur], ed[cur], bsBig);

    // out = relu(agg + h/deg + b)
    k_combine<<<dim3((unsigned)(((long)n * DD + 255) / 256), NB), 256, 0, stream>>>(
        BUF2, BUF1, deg, bv[l], n, bsBig);

    if (l < 3) {
      k_pnorm<<<1, 256, 0, stream>>>(ps[l], rnorm);
      k_score<<<dim3(n, NB), 256, 0, stream>>>(score, BUF2, ps[l], rnorm, bsBig);
      k_topk<<<NB, 512, 0, stream>>>(score, vals, perm, nidx, n, kout[l]);
      k_gather<<<dim3(kout[l], NB), 256, 0, stream>>>(BUF0, BUF2, perm, vals, bsSm, bsBig);
      k_remap<<<egrid, 256, 0, stream>>>(es[1 - cur], ed[1 - cur], em[1 - cur],
                                         es[cur], ed[cur], em[cur], nidx);
      cur = 1 - cur;
    }
  }

  // graph readout + MLP head
  k_mean<<<dim3(DD / 256, NB), 256, 0, stream>>>(g, BUF2, NK3, bsBig);
  k_fc<<<dim3(1024 / 256, NB), 256, 0, stream>>>(mh1, g,   lw1, lb1, DD,   1024, 0);
  k_fc<<<dim3(1024 / 256, NB), 256, 0, stream>>>(mh2, mh1, lw2, lb2, 1024, 1024, 0);
  k_fc<<<dim3(1, NB), 256, 0, stream>>>((float*)d_out, mh2, lw3, lb3, 1024, 200, 1);
}
